// CNV_Cifar10_Binary_Pad_23905787970192
// MI455X (gfx1250) — 8-bit matrix operand order repaired and checked by reading, compiled, not run on hardware
//
#include <hip/hip_runtime.h>
#include <hip/hip_bf16.h>
#include <stdint.h>

typedef __attribute__((ext_vector_type(8))) int v8i;

#define THREADS 256

__device__ __forceinline__ void async_copy16(const void* g, void* l) {
  uint64_t ga = (uint64_t)(size_t)g;
  uint32_t la = (uint32_t)(size_t)l;
  asm volatile("global_load_async_to_lds_b128 %0, %1, off" ::"v"(la), "v"(ga)
               : "memory");
}
__device__ __forceinline__ void async_wait0() {
  asm volatile("s_wait_asynccnt 0x0" ::: "memory");
}

__global__ void sign_f32_to_i8(const float* __restrict__ w, int8_t* __restrict__ o, long long n) {
  long long i = (long long)blockIdx.x * blockDim.x + threadIdx.x;
  if (i < n) {
    float v = w[i];
    o[i] = v > 0.f ? (int8_t)1 : (v < 0.f ? (int8_t)-1 : (int8_t)0);
  }
}

__global__ void conv1_f32(const float* __restrict__ x, const int8_t* __restrict__ w,
                          float* __restrict__ y, int Nimg) {
  long long idx = (long long)blockIdx.x * blockDim.x + threadIdx.x;
  long long total = (long long)Nimg * 64 * 32 * 32;
  if (idx >= total) return;
  int wcol = (int)(idx & 31);
  int hrow = (int)((idx >> 5) & 31);
  int co   = (int)((idx >> 10) & 63);
  int n    = (int)(idx >> 16);
  float acc = 0.f;
#pragma unroll
  for (int ci = 0; ci < 3; ++ci)
#pragma unroll
    for (int kh = 0; kh < 3; ++kh) {
      int ih = hrow + kh - 1;
      if ((unsigned)ih >= 32u) continue;
#pragma unroll
      for (int kw = 0; kw < 3; ++kw) {
        int iw = wcol + kw - 1;
        if ((unsigned)iw >= 32u) continue;
        float xv = x[(((long long)n * 3 + ci) * 32 + ih) * 32 + iw];
        acc += xv * (float)w[((co * 3 + ci) * 3 + kh) * 3 + kw];
      }
    }
  y[idx] = acc;
}

template <typename T>
__global__ void chan_stats(const T* __restrict__ x, int Nimg, int C, int HW,
                           float* __restrict__ mean, float* __restrict__ rstd) {
  int c = blockIdx.x;
  long long cnt = (long long)Nimg * HW;
  float s = 0.f, s2 = 0.f;
  for (long long i = threadIdx.x; i < cnt; i += blockDim.x) {
    long long n = i / HW, sp = i % HW;
    float v = (float)x[(n * C + c) * HW + sp];
    s += v;
    s2 += v * v;
  }
  __shared__ float sh0[THREADS];
  __shared__ float sh1[THREADS];
  sh0[threadIdx.x] = s;
  sh1[threadIdx.x] = s2;
  __syncthreads();
  for (int o = THREADS / 2; o > 0; o >>= 1) {
    if ((int)threadIdx.x < o) {
      sh0[threadIdx.x] += sh0[threadIdx.x + o];
      sh1[threadIdx.x] += sh1[threadIdx.x + o];
    }
    __syncthreads();
  }
  if (threadIdx.x == 0) {
    float inv = 1.f / (float)cnt;
    float m = sh0[0] * inv;
    float var = sh1[0] * inv - m * m;
    mean[c] = m;
    rstd[c] = rsqrtf(var + 1e-5f);
  }
}

template <typename T>
__global__ void bn_sign(const T* __restrict__ x, const float* __restrict__ mean,
                        const float* __restrict__ rstd, const float* __restrict__ gamma,
                        const float* __restrict__ beta, int8_t* __restrict__ o,
                        int C, int HW, long long total) {
  long long i = (long long)blockIdx.x * blockDim.x + threadIdx.x;
  if (i >= total) return;
  int c = (int)((i / HW) % C);
  float v = ((float)x[i] - mean[c]) * rstd[c] * gamma[c] + beta[c];
  o[i] = v > 0.f ? (int8_t)1 : (v < 0.f ? (int8_t)-1 : (int8_t)0);
}

__global__ void maxpool_i32(const int* __restrict__ x, int* __restrict__ y,
                            int Nimg, int C, int H, int W) {
  int Ho = H >> 1, Wo = W >> 1;
  long long tot = (long long)Nimg * C * Ho * Wo;
  long long i = (long long)blockIdx.x * blockDim.x + threadIdx.x;
  if (i >= tot) return;
  int wo = (int)(i % Wo);
  long long t = i / Wo;
  int ho = (int)(t % Ho);
  t /= Ho;
  int c = (int)(t % C);
  int n = (int)(t / C);
  const int* base = x + (((long long)(n * C + c) * H + 2 * ho) * W + 2 * wo);
  int a0 = base[0], a1 = base[1], a2 = base[W], a3 = base[W + 1];
  int m0 = a0 > a1 ? a0 : a1;
  int m1 = a2 > a3 ? a2 : a3;
  y[i] = m0 > m1 ? m0 : m1;
}

__global__ void __launch_bounds__(256)
binconv_wmma(const int8_t* __restrict__ act, const int8_t* __restrict__ wq,
             int* __restrict__ out, int Nimg, int Cin, int Cout, int H, int W) {
  const int K9 = Cin * 9;
  __shared__ __align__(16) int8_t As[32][64];
  __shared__ __align__(16) int8_t Bs[128][64];

  const int tid = threadIdx.x;
  const int lane = tid & 31;
  const int wave = tid >> 5;
  const int lhalf = lane >> 4;
  const int lmod = lane & 15;
  const int tm = (wave >> 2) * 16;
  const int tn = (wave & 3) * 16;
  const int mBase = blockIdx.y * 32;
  const long long pBase = (long long)blockIdx.x * 128;

  v8i c0 = {};
  v8i c1 = {};
  for (int k0 = 0; k0 < K9; k0 += 64) {
    if (k0 + 64 < K9)
      __builtin_prefetch(wq + (long long)(mBase + (tid & 31)) * K9 + k0 + 64, 0, 1);
    if (tid < 128) {
      int r = tid >> 2, q = (tid & 3) * 16;
      async_copy16(wq + (long long)(mBase + r) * K9 + k0 + q, &As[r][q]);
    }
    for (int idx = tid; idx < 128 * 64; idx += 256) {
      int nn = idx >> 6, kk = idx & 63;
      int k = k0 + kk;
      int8_t v = 0;
      long long p = pBase + nn;
      int ow = (int)(p % W);
      long long t = p / W;
      int oh = (int)(t % H);
      int img = (int)(t / H);
      int ci = k / 9, r9 = k % 9;
      int kh = r9 / 3, kw = r9 % 3;
      int ih = oh + kh - 1, iw = ow + kw - 1;
      if ((unsigned)ih < (unsigned)H && (unsigned)iw < (unsigned)W)
        v = act[((long long)(img * Cin + ci) * H + ih) * W + iw];
      Bs[nn][kk] = v;
    }
    async_wait0();
    __syncthreads();
    v8i a, b0, b1;
#pragma unroll
    for (int j = 0; j < 8; ++j) {
      int ka = (j & 3) * 4 + lhalf * 16 + (j >> 2) * 32;
      a[j] = *(const int*)&As[tm + lmod][ka];
      int kb = (j & 3) * 4 + lhalf * 16 + (j >> 2) * 32;
      b0[j] = *(const int*)&Bs[tn + lmod][kb];
      b1[j] = *(const int*)&Bs[64 + tn + lmod][kb];
    }
    c0 = __builtin_amdgcn_wmma_i32_16x16x64_iu8(true, a, true, b0, c0, false, false);
    c1 = __builtin_amdgcn_wmma_i32_16x16x64_iu8(true, a, true, b1, c1, false, false);
    __syncthreads();
  }
#pragma unroll
  for (int r = 0; r < 8; ++r) {
    int m = mBase + tm + r + 8 * lhalf;
#pragma unroll
    for (int half = 0; half < 2; ++half) {
      long long p = pBase + tn + lmod + 64 * half;
      int ow = (int)(p % W);
      long long t = p / W;
      int oh = (int)(t % H);
      int img = (int)(t / H);
      out[((long long)(img * Cout + m) * H + oh) * W + ow] = half ? c1[r] : c0[r];
    }
  }
}

__global__ void __launch_bounds__(256)
bingemm_wmma(const int8_t* __restrict__ A8, const int8_t* __restrict__ W8,
             int* __restrict__ out, int M, int Ncol, int K) {
  __shared__ __align__(16) int8_t As[32][64];
  __shared__ __align__(16) int8_t Bs[128][64];

  const int tid = threadIdx.x;
  const int lane = tid & 31;
  const int wave = tid >> 5;
  const int lhalf = lane >> 4;
  const int lmod = lane & 15;
  const int tm = (wave >> 2) * 16;
  const int tn = (wave & 3) * 16;
  const int mBase = blockIdx.y * 32;
  const int nBase = blockIdx.x * 128;

  v8i c0 = {};
  v8i c1 = {};
  for (int k0 = 0; k0 < K; k0 += 64) {
    if (k0 + 64 < K)
      __builtin_prefetch(A8 + (long long)(mBase + (tid & 31)) * K + k0 + 64, 0, 1);
    if (tid < 128) {
      int r = tid >> 2, q = (tid & 3) * 16;
      async_copy16(A8 + (long long)(mBase + r) * K + k0 + q, &As[r][q]);
    }
    for (int t2 = tid; t2 < 512; t2 += 256) {
      int r = t2 >> 2, q = (t2 & 3) * 16;
      async_copy16(W8 + (long long)(nBase + r) * K + k0 + q, &Bs[r][q]);
    }
    async_wait0();
    __syncthreads();
    v8i a, b0, b1;
#pragma unroll
    for (int j = 0; j < 8; ++j) {
      int ka = (j & 3) * 4 + lhalf * 16 + (j >> 2) * 32;
      a[j] = *(const int*)&As[tm + lmod][ka];
      int kb = (j & 3) * 4 + lhalf * 16 + (j >> 2) * 32;
      b0[j] = *(const int*)&Bs[tn + lmod][kb];
      b1[j] = *(const int*)&Bs[64 + tn + lmod][kb];
    }
    c0 = __builtin_amdgcn_wmma_i32_16x16x64_iu8(true, a, true, b0, c0, false, false);
    c1 = __builtin_amdgcn_wmma_i32_16x16x64_iu8(true, a, true, b1, c1, false, false);
    __syncthreads();
  }
#pragma unroll
  for (int r = 0; r < 8; ++r) {
    int m = mBase + tm + r + 8 * lhalf;
    int n0 = nBase + tn + lmod;
    out[(long long)m * Ncol + n0] = c0[r];
    out[(long long)m * Ncol + n0 + 64] = c1[r];
  }
}

__global__ void fc3_i32(const int8_t* __restrict__ A8, const int8_t* __restrict__ W8,
                        int* __restrict__ out) {
  int i = blockIdx.x * blockDim.x + threadIdx.x;
  if (i >= 512 * 10) return;
  int j = i % 10, row = i / 10;
  int acc = 0;
  for (int k = 0; k < 512; ++k) acc += (int)A8[row * 512 + k] * (int)W8[j * 512 + k];
  out[i] = acc;
}

__global__ void bn_logsoftmax(const int* __restrict__ logits, const float* __restrict__ mean,
                              const float* __restrict__ rstd, float* __restrict__ out) {
  int row = blockIdx.x * blockDim.x + threadIdx.x;
  if (row >= 512) return;
  float v[10];
  float mx = -1e30f;
#pragma unroll
  for (int j = 0; j < 10; ++j) {
    v[j] = ((float)logits[row * 10 + j] - mean[j]) * rstd[j];
    mx = fmaxf(mx, v[j]);
  }
  float s = 0.f;
#pragma unroll
  for (int j = 0; j < 10; ++j) s += __expf(v[j] - mx);
  float lse = __logf(s) + mx;
#pragma unroll
  for (int j = 0; j < 10; ++j) out[row * 10 + j] = v[j] - lse;
}

extern "C" void kernel_launch(void* const* d_in, const int* in_sizes, int n_in,
                              void* d_out, int out_size, void* d_ws, size_t ws_size,
                              hipStream_t stream) {
  (void)in_sizes; (void)n_in; (void)out_size; (void)ws_size;
  const int NIMG = 512;
  char* ws = (char*)d_ws;
  const float* x = (const float*)d_in[0];
  const float* wsrc[9] = {
      (const float*)d_in[1],  (const float*)d_in[4],  (const float*)d_in[7],
      (const float*)d_in[10], (const float*)d_in[13], (const float*)d_in[16],
      (const float*)d_in[19], (const float*)d_in[22], (const float*)d_in[25]};
  const long long wcnt[9] = {64 * 3 * 9,    64 * 64 * 9,   128 * 64 * 9,
                             128 * 128 * 9, 256 * 128 * 9, 256 * 256 * 9,
                             512LL * 4096,  512LL * 512,   10LL * 512};

  size_t off = 0;
  int8_t* w8[9];
  for (int i = 0; i < 9; ++i) { w8[i] = (int8_t*)(ws + off); off += (size_t)wcnt[i]; }
  off = (off + 255) & ~(size_t)255;
  float* meanb = (float*)(ws + off); off += 2048;
  float* rstdb = (float*)(ws + off); off += 2048;
  off = (off + 255) & ~(size_t)255;
  char* ACC = ws + off;  off += (size_t)136 << 20;
  char* POOL = ws + off; off += (size_t)34 << 20;
  int8_t* ACT0 = (int8_t*)(ws + off); off += (size_t)34 << 20;
  int8_t* ACT1 = (int8_t*)(ws + off);

  auto cdiv = [](long long a, long long b) { return (unsigned)((a + b - 1) / b); };

  for (int i = 0; i < 9; ++i)
    sign_f32_to_i8<<<cdiv(wcnt[i], THREADS), THREADS, 0, stream>>>(wsrc[i], w8[i], wcnt[i]);

  {
    long long tot = (long long)NIMG * 64 * 1024;
    conv1_f32<<<cdiv(tot, THREADS), THREADS, 0, stream>>>(x, w8[0], (float*)ACC, NIMG);
    chan_stats<float><<<64, THREADS, 0, stream>>>((const float*)ACC, NIMG, 64, 1024, meanb, rstdb);
    bn_sign<float><<<cdiv(tot, THREADS), THREADS, 0, stream>>>(
        (const float*)ACC, meanb, rstdb, (const float*)d_in[2], (const float*)d_in[3],
        ACT0, 64, 1024, tot);
  }
  {
    dim3 g(cdiv((long long)NIMG * 1024, 128), 64 / 32);
    binconv_wmma<<<g, THREADS, 0, stream>>>(ACT0, w8[1], (int*)ACC, NIMG, 64, 64, 32, 32);
    long long pt = (long long)NIMG * 64 * 256;
    maxpool_i32<<<cdiv(pt, THREADS), THREADS, 0, stream>>>((const int*)ACC, (int*)POOL, NIMG, 64, 32, 32);
    chan_stats<int><<<64, THREADS, 0, stream>>>((const int*)POOL, NIMG, 64, 256, meanb, rstdb);
    bn_sign<int><<<cdiv(pt, THREADS), THREADS, 0, stream>>>(
        (const int*)POOL, meanb, rstdb, (const float*)d_in[5], (const float*)d_in[6],
        ACT1, 64, 256, pt);
  }
  {
    dim3 g(cdiv((long long)NIMG * 256, 128), 128 / 32);
    binconv_wmma<<<g, THREADS, 0, stream>>>(ACT1, w8[2], (int*)ACC, NIMG, 64, 128, 16, 16);
    long long tot = (long long)NIMG * 128 * 256;
    chan_stats<int><<<128, THREADS, 0, stream>>>((const int*)ACC, NIMG, 128, 256, meanb, rstdb);
    bn_sign<int><<<cdiv(tot, THREADS), THREADS, 0, stream>>>(
        (const int*)ACC, meanb, rstdb, (const float*)d_in[8], (const float*)d_in[9],
        ACT0, 128, 256, tot);
  }
  {
    dim3 g(cdiv((long long)NIMG * 256, 128), 128 / 32);
    binconv_wmma<<<g, THREADS, 0, stream>>>(ACT0, w8[3], (int*)ACC, NIMG, 128, 128, 16, 16);
    long long pt = (long long)NIMG * 128 * 64;
    maxpool_i32<<<cdiv(pt, THREADS), THREADS, 0, stream>>>((const int*)ACC, (int*)POOL, NIMG, 128, 16, 16);
    chan_stats<int><<<128, THREADS, 0, stream>>>((const int*)POOL, NIMG, 128, 64, meanb, rstdb);
    bn_sign<int><<<cdiv(pt, THREADS), THREADS, 0, stream>>>(
        (const int*)POOL, meanb, rstdb, (const float*)d_in[11], (const float*)d_in[12],
        ACT1, 128, 64, pt);
  }
  {
    dim3 g(cdiv((long long)NIMG * 64, 128), 256 / 32);
    binconv_wmma<<<g, THREADS, 0, stream>>>(ACT1, w8[4], (int*)ACC, NIMG, 128, 256, 8, 8);
    long long tot = (long long)NIMG * 256 * 64;
    chan_stats<int><<<256, THREADS, 0, stream>>>((const int*)ACC, NIMG, 256, 64, meanb, rstdb);
    bn_sign<int><<<cdiv(tot, THREADS), THREADS, 0, stream>>>(
        (const int*)ACC, meanb, rstdb, (const float*)d_in[14], (const float*)d_in[15],
        ACT0, 256, 64, tot);
  }
  {
    dim3 g(cdiv((long long)NIMG * 64, 128), 256 / 32);
    binconv_wmma<<<g, THREADS, 0, stream>>>(ACT0, w8[5], (int*)ACC, NIMG, 256, 256, 8, 8);
    long long pt = (long long)NIMG * 256 * 16;
    maxpool_i32<<<cdiv(pt, THREADS), THREADS, 0, stream>>>((const int*)ACC, (int*)POOL, NIMG, 256, 8, 8);
    chan_stats<int><<<256, THREADS, 0, stream>>>((const int*)POOL, NIMG, 256, 16, meanb, rstdb);
    bn_sign<int><<<cdiv(pt, THREADS), THREADS, 0, stream>>>(
        (const int*)POOL, meanb, rstdb, (const float*)d_in[17], (const float*)d_in[18],
        ACT1, 256, 16, pt);
  }
  {
    dim3 g(512 / 128, 512 / 32);
    bingemm_wmma<<<g, THREADS, 0, stream>>>(ACT1, w8[6], (int*)ACC, 512, 512, 4096);
    long long tot = 512LL * 512;
    chan_stats<int><<<512, THREADS, 0, stream>>>((const int*)ACC, 512, 512, 1, meanb, rstdb);
    bn_sign<int><<<cdiv(tot, THREADS), THREADS, 0, stream>>>(
        (const int*)ACC, meanb, rstdb, (const float*)d_in[20], (const float*)d_in[21],
        ACT0, 512, 1, tot);
  }
  {
    dim3 g(512 / 128, 512 / 32);
    bingemm_wmma<<<g, THREADS, 0, stream>>>(ACT0, w8[7], (int*)ACC, 512, 512, 512);
    long long tot = 512LL * 512;
    chan_stats<int><<<512, THREADS, 0, stream>>>((const int*)ACC, 512, 512, 1, meanb, rstdb);
    bn_sign<int><<<cdiv(tot, THREADS), THREADS, 0, stream>>>(
        (const int*)ACC, meanb, rstdb, (const float*)d_in[23], (const float*)d_in[24],
        ACT1, 512, 1, tot);
  }
  {
    int* logits = (int*)POOL;
    fc3_i32<<<cdiv(512 * 10, THREADS), THREADS, 0, stream>>>(ACT1, w8[8], logits);
    chan_stats<int><<<10, THREADS, 0, stream>>>((const int*)logits, 512, 10, 1, meanb, rstdb);
    bn_logsoftmax<<<cdiv(512, THREADS), THREADS, 0, stream>>>(logits, meanb, rstdb, (float*)d_out);
  }
}
